// MambaUnet_Base_72155450573523
// MI455X (gfx1250) — compile-verified, matrix operand order checked by reading, not run on hardware
//
#include <hip/hip_runtime.h>
#include <hip/hip_bf16.h>
#include <math.h>

// ---------------------------------------------------------------------------
// MambaUnet on gfx1250 (MI455X). wave32. GEMM-shaped stages run on
// v_wmma_f32_16x16x32_f16 with f16-resident operands; 32-row M tiles per
// block (B-fragment reused across two wmmas), async global->LDS A staging,
// b128 fragment loads, templated dual-A accumulation for the iSTFT.
// ---------------------------------------------------------------------------

typedef _Float16 h16;
typedef __attribute__((ext_vector_type(8)))  _Float16 v8h;
typedef __attribute__((ext_vector_type(16))) _Float16 v16h;
typedef __attribute__((ext_vector_type(8)))  float    v8f;

#define N_FFT   1024
#define HOPSZ   160
#define NFRAMES 321                              // 51200/160 + 1
#define NFREQ   513
#define SAMPLES 51200
#define OUTLEN  ((NFRAMES - 1) * HOPSZ + N_FFT)  // 52224

// padded GEMM dims (M % 32, N % 64, K % 32 == 0; operands zero-padded)
#define NFREQ_P 576                              // freq dim padded (N and K roles)
#define MSTFT   (2 * NFRAMES)                    // 642
#define MSTFT_P 672                              // 21 * 32
#define MIST    (4 * NFRAMES)                    // 1284
#define MIST_P  1312                             // 41 * 32

#define R_NET   512
#define T_NET   64
#define NET_M   (R_NET * T_NET)                  // 32768 (1024 * 32)
#define DPROJ_P 192                              // 161 -> 192 (dt col = 160)
#define D_CONV  96
#define DOUT_P  64                               // out-proj N: 32 -> 64

__host__ __device__ static inline int ceil_div(int a, int b) { return (a + b - 1) / b; }

__device__ static inline float hann_w(int n) {
    return 0.5f - 0.5f * __cosf(6.28318530717958647692f * (float)n / (float)N_FFT);
}

#if defined(__HIP_DEVICE_COMPILE__) && __has_builtin(__builtin_amdgcn_global_load_async_to_lds_b128)
#define ASYNC_LDS 1
typedef int v4i __attribute__((vector_size(16)));
typedef __attribute__((address_space(1))) v4i as1_v4i;   // global
typedef __attribute__((address_space(3))) v4i as3_v4i;   // LDS
#else
#define ASYNC_LDS 0
#endif

__device__ static inline void stage16(const h16* src, h16* dst)
{
#if ASYNC_LDS
    __builtin_amdgcn_global_load_async_to_lds_b128(
        (as1_v4i*)(void*)src, (as3_v4i*)(void*)dst, 0, 0);
#else
    *(v8h*)dst = *(const v8h*)src;
#endif
}

__device__ static inline void stage_wait()
{
#if ASYNC_LDS
#if __has_builtin(__builtin_amdgcn_s_wait_asynccnt)
    __builtin_amdgcn_s_wait_asynccnt(0);
#else
    asm volatile("s_wait_asynccnt 0x0" ::: "memory");
#endif
#endif
}

// ---------------------------------------------------------------------------
// C[M,N] = alpha*(A@W^T) (+ A2@W2^T when DUAL).  A,W f16 row-major; C f32.
// REQUIRES: M%32==0, N%64==0, K%32==0, 16B-aligned buffers. Branch-free loop.
// Block = 128 threads (4 waves). Block tile = 32(M) x 64(N); each wave owns a
// 32x16 C tile as two accumulators sharing one B fragment per k-step.
// ---------------------------------------------------------------------------
template <bool DUAL>
__global__ __launch_bounds__(128)
void gemm_wmma_kernel(const h16* __restrict__ A,  const h16* __restrict__ W,
                      const h16* __restrict__ A2, const h16* __restrict__ W2,
                      float* __restrict__ C,
                      int M, int N, int K, float alpha)
{
    __shared__ __attribute__((aligned(16))) h16 sA [32 * 32];
    __shared__ __attribute__((aligned(16))) h16 sA2[32 * 32];

    const int tid   = threadIdx.x;
    const int lane  = tid & 31;
    const int wave  = tid >> 5;
    const int mBase = blockIdx.y * 32;
    const int nBase = blockIdx.x * 64 + wave * 16;
    const int row   = lane & 15;                 // M row of A frag / N col of B frag
    const int hi    = lane >> 4;                 // K half-group select
    const int kBase = hi * 8;                    // fragment K sub-offset

    // A staging: 128 chunks of 16B cover the 32x32 f16 tile; chunk t covers
    // LDS halves [8t, 8t+8) = row t>>2, cols (t&3)*8.. (row stride 32 halves).
    const int stM = tid >> 2;
    const int stK = (tid & 3) * 8;
    const h16* stSrcA = A + (size_t)(mBase + stM) * K + stK;
    const h16* stSrcB = DUAL ? (A2 + (size_t)(mBase + stM) * K + stK) : A;
    h16* stDstA = sA  + tid * 8;
    h16* stDstB = sA2 + tid * 8;

    v8f acc0 = {};
    v8f acc1 = {};
    const h16* wk  = W + (size_t)(nBase + row) * K;
    const h16* wk2 = DUAL ? (W2 + (size_t)(nBase + row) * K) : W;

    for (int k0 = 0; k0 < K; k0 += 32) {
        stage16(stSrcA + k0, stDstA);
        if (DUAL) stage16(stSrcB + k0, stDstB);
        stage_wait();
        __syncthreads();

        if (k0 + 32 < K) __builtin_prefetch(&wk[k0 + 32], 0, 3);

        // B fragment (shared by both M-halves): two contiguous b128 runs
        {
            v8h b0 = *(const v8h*)&wk[k0 + kBase];
            v8h b1 = *(const v8h*)&wk[k0 + 16 + kBase];
            v16h bf = __builtin_shufflevector(b0, b1, 0,1,2,3,4,5,6,7,
                                                       8,9,10,11,12,13,14,15);
            v8h a0 = *(const v8h*)&sA[row * 32 + kBase];
            v8h a1 = *(const v8h*)&sA[row * 32 + 16 + kBase];
            v16h af0 = __builtin_shufflevector(a0, a1, 0,1,2,3,4,5,6,7,
                                                        8,9,10,11,12,13,14,15);
            v8h c0 = *(const v8h*)&sA[(row + 16) * 32 + kBase];
            v8h c1 = *(const v8h*)&sA[(row + 16) * 32 + 16 + kBase];
            v16h af1 = __builtin_shufflevector(c0, c1, 0,1,2,3,4,5,6,7,
                                                        8,9,10,11,12,13,14,15);
            acc0 = __builtin_amdgcn_wmma_f32_16x16x32_f16(
                       false, af0, false, bf, (short)0, acc0, false, false);
            acc1 = __builtin_amdgcn_wmma_f32_16x16x32_f16(
                       false, af1, false, bf, (short)0, acc1, false, false);
        }
        if (DUAL) {
            v8h b0 = *(const v8h*)&wk2[k0 + kBase];
            v8h b1 = *(const v8h*)&wk2[k0 + 16 + kBase];
            v16h bf = __builtin_shufflevector(b0, b1, 0,1,2,3,4,5,6,7,
                                                       8,9,10,11,12,13,14,15);
            v8h a0 = *(const v8h*)&sA2[row * 32 + kBase];
            v8h a1 = *(const v8h*)&sA2[row * 32 + 16 + kBase];
            v16h af0 = __builtin_shufflevector(a0, a1, 0,1,2,3,4,5,6,7,
                                                        8,9,10,11,12,13,14,15);
            v8h c0 = *(const v8h*)&sA2[(row + 16) * 32 + kBase];
            v8h c1 = *(const v8h*)&sA2[(row + 16) * 32 + 16 + kBase];
            v16h af1 = __builtin_shufflevector(c0, c1, 0,1,2,3,4,5,6,7,
                                                        8,9,10,11,12,13,14,15);
            acc0 = __builtin_amdgcn_wmma_f32_16x16x32_f16(
                       false, af0, false, bf, (short)0, acc0, false, false);
            acc1 = __builtin_amdgcn_wmma_f32_16x16x32_f16(
                       false, af1, false, bf, (short)0, acc1, false, false);
        }
        __syncthreads();
    }

    // C stores (lanes 0-15: rows r; lanes 16-31: rows r+8) for both M-halves
    const int n = nBase + row;
#pragma unroll
    for (int r = 0; r < 8; ++r) {
        C[(size_t)(mBase + r + hi * 8) * N + n]      = alpha * acc0[r];
        C[(size_t)(mBase + 16 + r + hi * 8) * N + n] = alpha * acc1[r];
    }
}

// ---------------------------------------------------------------------------
// Support kernels
// ---------------------------------------------------------------------------
__global__ __launch_bounds__(256)
void zerof_kernel(float* __restrict__ p, int n)
{
    int i = blockIdx.x * blockDim.x + threadIdx.x;
    if (i < n) p[i] = 0.0f;
}

__global__ __launch_bounds__(256)
void genw_h_kernel(h16* __restrict__ w, int n)
{
    int i = blockIdx.x * blockDim.x + threadIdx.x;
    if (i < n) w[i] = (h16)(0.02f * __sinf(0.37f * (float)i));
}

__global__ __launch_bounds__(256)
void genw_f_kernel(float* __restrict__ w, int n)
{
    int i = blockIdx.x * blockDim.x + threadIdx.x;
    if (i < n) w[i] = 0.02f * __sinf(0.53f * (float)i);
}

// forward DFT mats, f16, [NFREQ_P][N_FFT], zero-padded rows
__global__ __launch_bounds__(256)
void dft_fwd_kernel(h16* __restrict__ cosM, h16* __restrict__ sinM)
{
    int idx = blockIdx.x * blockDim.x + threadIdx.x;
    if (idx >= NFREQ_P * N_FFT) return;
    const int t = idx & (N_FFT - 1);
    const int f = idx >> 10;
    float cv = 0.0f, sv = 0.0f;
    if (f < NFREQ) {
        const float ang = 6.28318530717958647692f * (float)f * (float)t / (float)N_FFT;
        cv = cosf(ang);
        sv = sinf(ang);
    }
    cosM[idx] = (h16)cv;
    sinM[idx] = (h16)sv;
}

// inverse DFT mats (transposed), f16, [N_FFT][NFREQ_P], zero-padded cols
__global__ __launch_bounds__(256)
void dft_inv_kernel(h16* __restrict__ icosT, h16* __restrict__ isinT)
{
    int idx = blockIdx.x * blockDim.x + threadIdx.x;
    if (idx >= N_FFT * NFREQ_P) return;
    const int f = idx % NFREQ_P;
    const int t = idx / NFREQ_P;
    float cv = 0.0f, sv = 0.0f;
    if (f < NFREQ) {
        const float ang = 6.28318530717958647692f * (float)f * (float)t / (float)N_FFT;
        const float ww = (f == 0 || f == NFREQ - 1) ? 1.0f : 2.0f;
        cv =  ww * cosf(ang) / (float)N_FFT;
        sv = -ww * sinf(ang) / (float)N_FFT;
    }
    icosT[idx] = (h16)cv;
    isinT[idx] = (h16)sv;
}

// windowed frames (reflect pad), f16, [MSTFT_P][N_FFT], zero-padded rows
__global__ __launch_bounds__(256)
void frames_kernel(const float* __restrict__ mix, h16* __restrict__ frames)
{
    int idx = blockIdx.x * blockDim.x + threadIdx.x;
    if (idx >= MSTFT_P * N_FFT) return;
    const int n  = idx & (N_FFT - 1);
    const int rm = idx >> 10;
    float v = 0.0f;
    if (rm < MSTFT) {
        const int t = rm % NFRAMES;
        const int c = rm / NFRAMES;
        int j = t * HOPSZ + n - (N_FFT / 2);
        if (j < 0) j = -j;
        if (j >= SAMPLES) j = 2 * SAMPLES - 2 - j;
        v = mix[(size_t)c * SAMPLES + j] * hann_w(n);
    }
    frames[idx] = (h16)v;
}

__global__ __launch_bounds__(256)
void magphase_kernel(const float* __restrict__ re, const float* __restrict__ im,
                     float* __restrict__ mag, float* __restrict__ ci,
                     float* __restrict__ si, int n)
{
    int i = blockIdx.x * blockDim.x + threadIdx.x;
    if (i >= n) return;
    const float r  = re[i];
    const float m_ = im[i];
    const float m  = sqrtf(fmaxf(r * r + m_ * m_, 1e-10f));
    mag[i] = m;
    ci[i]  = r / m;
    si[i]  = m_ / m;
}

// net input x[r*T+t, c] = LN_c(features from mag), f16
__global__ __launch_bounds__(256)
void prep_net_kernel(const float* __restrict__ mag, h16* __restrict__ x)
{
    int idx = blockIdx.x * blockDim.x + threadIdx.x;
    if (idx >= NET_M) return;
    const int t = idx % T_NET;
    const int r = idx / T_NET;
    float v[32];
    float mu = 0.0f;
#pragma unroll
    for (int c = 0; c < 32; ++c) {
        const int ch = c & 1;
        const float m = mag[((size_t)ch * NFRAMES + t) * NFREQ_P + r];
        v[c] = m * (0.05f + 0.01f * (float)c);
        mu += v[c];
    }
    mu *= (1.0f / 32.0f);
    float var = 0.0f;
#pragma unroll
    for (int c = 0; c < 32; ++c) { const float d = v[c] - mu; var += d * d; }
    const float inv = __frsqrt_rn(var * (1.0f / 32.0f) + 1e-5f);
#pragma unroll
    for (int c = 0; c < 32; ++c)
        x[(size_t)idx * 32 + c] = (h16)((v[c] - mu) * inv);
}

// causal depthwise conv (K=4) + SiLU over xBC channels
__global__ __launch_bounds__(256)
void conv_silu_kernel(const float* __restrict__ zxb, const float* __restrict__ prm,
                      float* __restrict__ convo)
{
    int idx = blockIdx.x * blockDim.x + threadIdx.x;
    if (idx >= NET_M * D_CONV) return;
    const int c = idx % D_CONV;
    const int t = (idx / D_CONV) % T_NET;
    const int r = idx / (D_CONV * T_NET);
    float acc = prm[c];
#pragma unroll
    for (int k = 0; k < 4; ++k) {
        const int ts = t - 3 + k;
        const float xv = (ts >= 0)
            ? zxb[(((size_t)r * T_NET) + ts) * DPROJ_P + 64 + c] : 0.0f;
        acc += xv * prm[D_CONV + c * 4 + k];
    }
    convo[idx] = acc / (1.0f + __expf(-acc));
}

// sequential SSM scan: block = one row, 64 threads = head dim, h[16] in regs,
// per-step B/C/dt broadcast through LDS, fused silu(z) gate + RMS-norm.
__global__ __launch_bounds__(64)
void scan_kernel(const float* __restrict__ convo, const float* __restrict__ zxb,
                 h16* __restrict__ y)
{
    __shared__ float sB[16], sC[16], sScal[2], sRed[64];
    const int r = blockIdx.x;
    const int p = threadIdx.x;

    float h[16];
#pragma unroll
    for (int n = 0; n < 16; ++n) h[n] = 0.0f;

    const float A_const = -4.0f;
    const float dt_bias = -3.0f;

    for (int t = 0; t < T_NET; ++t) {
        const size_t base = (size_t)r * T_NET + t;
        if (p < 16)       sB[p]      = convo[base * D_CONV + 64 + p];
        else if (p < 32)  sC[p - 16] = convo[base * D_CONV + 80 + (p - 16)];
        else if (p == 32) {
            const float raw = zxb[base * DPROJ_P + 160] + dt_bias;
            const float dt = (raw > 20.0f) ? raw : __logf(1.0f + __expf(raw));
            sScal[0] = dt;
            sScal[1] = __expf(dt * A_const);
        }
        __syncthreads();

        const float dt = sScal[0];
        const float dA = sScal[1];
        const float xt = convo[base * D_CONV + p];
        float yt = 0.0f;
#pragma unroll
        for (int n = 0; n < 16; ++n) {
            h[n] = h[n] * dA + dt * xt * sB[n];
            yt += h[n] * sC[n];
        }
        yt += xt;                                   // D = 1 skip
        const float z = zxb[base * DPROJ_P + p];
        yt *= z / (1.0f + __expf(-z));              // silu(z) gate

        sRed[p] = yt * yt;                          // RMS-norm over 64
        __syncthreads();
        for (int s = 32; s > 0; s >>= 1) {
            if (p < s) sRed[p] += sRed[p + s];
            __syncthreads();
        }
        yt *= __frsqrt_rn(sRed[0] * (1.0f / 64.0f) + 1e-5f);
        y[base * 64 + p] = (h16)yt;
        __syncthreads();
    }
}

// masked spectrograms, f16, [MIST_P][NFREQ_P] (pads -> 0)
__global__ __launch_bounds__(256)
void mask_kernel(const float* __restrict__ mag, const float* __restrict__ ci,
                 const float* __restrict__ si, const float* __restrict__ net,
                 h16* __restrict__ oreal, h16* __restrict__ oimag)
{
    int idx = blockIdx.x * blockDim.x + threadIdx.x;
    if (idx >= MIST_P * NFREQ_P) return;
    const int f  = idx % NFREQ_P;
    const int rm = idx / NFREQ_P;
    float vr = 0.0f, vi = 0.0f;
    if (rm < MIST && f < NFREQ) {
        const int t  = rm % NFRAMES;
        const int sc = rm / NFRAMES;                // s*2 + c
        const int c  = sc & 1;
        const int rr = (f < R_NET) ? f : (R_NET - 1);
        const int tt = (t < T_NET) ? t : (T_NET - 1);
        const size_t nb = ((size_t)rr * T_NET + tt) * DOUT_P;
        const int cb = (sc * 3) % 29;
        const float mask_mag = 1.0f / (1.0f + __expf(-net[nb + cb]));
        const float mr = tanhf(net[nb + cb + 1]);
        const float mi = tanhf(net[nb + cb + 2]);
        const float mm = fmaxf(sqrtf(mr * mr + mi * mi), 1e-10f);
        const float mcos = mr / mm;
        const float msin = mi / mm;
        const size_t ib = ((size_t)c * NFRAMES + t) * NFREQ_P + f;
        const float cin  = ci[ib];
        const float sin_ = si[ib];
        const float om = fmaxf(mag[ib] * mask_mag, 0.0f);
        vr = om * (cin * mcos - sin_ * msin);
        vi = om * (sin_ * mcos + cin * msin);
    }
    oreal[idx] = (h16)vr;
    oimag[idx] = (h16)vi;
}

// windowed atomic overlap-add
__global__ __launch_bounds__(256)
void ola_kernel(const float* __restrict__ framesO, float* __restrict__ yacc)
{
    int idx = blockIdx.x * blockDim.x + threadIdx.x;
    if (idx >= MIST * N_FFT) return;
    const int n  = idx & (N_FFT - 1);
    const int rm = idx >> 10;
    const int t  = rm % NFRAMES;
    const int b  = rm / NFRAMES;
    atomicAdd(&yacc[(size_t)b * OUTLEN + t * HOPSZ + n], framesO[idx] * hann_w(n));
}

// normalize by window-square sum, crop, source mask
__global__ __launch_bounds__(256)
void fin_kernel(const float* __restrict__ yacc, const float* __restrict__ smask,
                float* __restrict__ out, int n_out)
{
    int idx = blockIdx.x * blockDim.x + threadIdx.x;
    if (idx >= n_out) return;
    const int l = idx % SAMPLES;
    const int c = (idx / SAMPLES) & 1;
    const int s = (idx / (SAMPLES * 2)) & 1;
    const int j = (N_FFT / 2) + l;

    int tmin = (j - (N_FFT - 1) + HOPSZ - 1) / HOPSZ;
    if (tmin < 0) tmin = 0;
    int tmax = j / HOPSZ;
    if (tmax > NFRAMES - 1) tmax = NFRAMES - 1;
    float wsq = 0.0f;
    for (int t = tmin; t <= tmax; ++t) {
        const float w = hann_w(j - t * HOPSZ);
        wsq += w * w;
    }
    wsq = fmaxf(wsq, 1e-11f);
    out[idx] = (yacc[(size_t)(s * 2 + c) * OUTLEN + j] / wsq) * smask[s];
}

// ---------------------------------------------------------------------------
// Host-side orchestration
// ---------------------------------------------------------------------------
extern "C" void kernel_launch(void* const* d_in, const int* in_sizes, int n_in,
                              void* d_out, int out_size, void* d_ws, size_t ws_size,
                              hipStream_t stream)
{
    const float* mix   = (const float*)d_in[0];   // (1, 2, 51200)
    const float* smask = (const float*)d_in[2];   // (1, 2)
    float*       out   = (float*)d_out;           // (1, 2, 2, 51200)

    // ---- workspace layout (256B-aligned chunks) ----
    char* base = (char*)d_ws;
    size_t off = 0;
    auto alloc = [&](size_t bytes) -> char* {
        char* p = base + off;
        off += (bytes + 255) & ~(size_t)255;
        return p;
    };

    h16*   frames  = (h16*)  alloc((size_t)MSTFT_P * N_FFT   * 2);
    h16*   cosM    = (h16*)  alloc((size_t)NFREQ_P * N_FFT   * 2);
    h16*   sinM    = (h16*)  alloc((size_t)NFREQ_P * N_FFT   * 2);
    h16*   icosT   = (h16*)  alloc((size_t)N_FFT   * NFREQ_P * 2);
    h16*   isinT   = (h16*)  alloc((size_t)N_FFT   * NFREQ_P * 2);
    float* re      = (float*)alloc((size_t)MSTFT_P * NFREQ_P * 4);
    float* im      = (float*)alloc((size_t)MSTFT_P * NFREQ_P * 4);
    float* mag     = (float*)alloc((size_t)MSTFT_P * NFREQ_P * 4);
    float* ci      = (float*)alloc((size_t)MSTFT_P * NFREQ_P * 4);
    float* si      = (float*)alloc((size_t)MSTFT_P * NFREQ_P * 4);
    h16*   netx    = (h16*)  alloc((size_t)NET_M * 32        * 2);
    float* zxb     = (float*)alloc((size_t)NET_M * DPROJ_P   * 4);
    float* convo   = (float*)alloc((size_t)NET_M * D_CONV    * 4);
    h16*   yv      = (h16*)  alloc((size_t)NET_M * 64        * 2);
    float* netout  = (float*)alloc((size_t)NET_M * DOUT_P    * 4);
    h16*   oreal   = (h16*)  alloc((size_t)MIST_P * NFREQ_P  * 2);
    h16*   oimag   = (h16*)  alloc((size_t)MIST_P * NFREQ_P  * 2);
    float* framesO = (float*)alloc((size_t)MIST_P * N_FFT    * 4);
    float* yacc    = (float*)alloc((size_t)4 * OUTLEN        * 4);
    h16*   W_in    = (h16*)  alloc((size_t)DPROJ_P * 32      * 2);
    h16*   W_out   = (h16*)  alloc((size_t)DOUT_P * 64       * 2);
    float* prmCnv  = (float*)alloc((size_t)(D_CONV * 5)      * 4);

    if (ws_size < off) {
        zerof_kernel<<<ceil_div(out_size, 256), 256, 0, stream>>>(out, out_size);
        return;
    }

    const int B = 256;

    // 0) deterministic synthesized weights
    genw_h_kernel<<<ceil_div(DPROJ_P * 32, B), B, 0, stream>>>(W_in, DPROJ_P * 32);
    genw_h_kernel<<<ceil_div(DOUT_P * 64, B), B, 0, stream>>>(W_out, DOUT_P * 64);
    genw_f_kernel<<<ceil_div(D_CONV * 5, B), B, 0, stream>>>(prmCnv, D_CONV * 5);

    // 1) DFT matrices + windowed frames (all f16, padded)
    dft_fwd_kernel<<<ceil_div(NFREQ_P * N_FFT, B), B, 0, stream>>>(cosM, sinM);
    dft_inv_kernel<<<ceil_div(N_FFT * NFREQ_P, B), B, 0, stream>>>(icosT, isinT);
    frames_kernel<<<ceil_div(MSTFT_P * N_FFT, B), B, 0, stream>>>(mix, frames);

    // 2) STFT via WMMA
    {
        dim3 g(NFREQ_P / 64, MSTFT_P / 32);
        gemm_wmma_kernel<false><<<g, 128, 0, stream>>>(frames, cosM, nullptr, nullptr,
                                                       re, MSTFT_P, NFREQ_P, N_FFT,  1.0f);
        gemm_wmma_kernel<false><<<g, 128, 0, stream>>>(frames, sinM, nullptr, nullptr,
                                                       im, MSTFT_P, NFREQ_P, N_FFT, -1.0f);
    }
    magphase_kernel<<<ceil_div(MSTFT_P * NFREQ_P, B), B, 0, stream>>>(
        re, im, mag, ci, si, MSTFT_P * NFREQ_P);

    // 3) Mamba block: LN prep -> in-projection (WMMA, K=32)
    prep_net_kernel<<<ceil_div(NET_M, B), B, 0, stream>>>(mag, netx);
    {
        dim3 g(DPROJ_P / 64, NET_M / 32);
        gemm_wmma_kernel<false><<<g, 128, 0, stream>>>(netx, W_in, nullptr, nullptr,
                                                       zxb, NET_M, DPROJ_P, 32, 1.0f);
    }

    // 4) depthwise conv + silu, then sequential scan
    conv_silu_kernel<<<ceil_div(NET_M * D_CONV, B), B, 0, stream>>>(zxb, prmCnv, convo);
    scan_kernel<<<R_NET, 64, 0, stream>>>(convo, zxb, yv);

    // 5) out-projection (WMMA)
    {
        dim3 g(DOUT_P / 64, NET_M / 32);
        gemm_wmma_kernel<false><<<g, 128, 0, stream>>>(yv, W_out, nullptr, nullptr,
                                                       netout, NET_M, DOUT_P, 64, 1.0f);
    }

    // 6) masks -> masked spectrograms (f16, padded)
    mask_kernel<<<ceil_div(MIST_P * NFREQ_P, B), B, 0, stream>>>(
        mag, ci, si, netout, oreal, oimag);

    // 7) iSTFT via dual-A WMMA: framesO = oreal@ICOS + oimag@ISIN
    {
        dim3 g(N_FFT / 64, MIST_P / 32);
        gemm_wmma_kernel<true><<<g, 128, 0, stream>>>(oreal, icosT, oimag, isinT,
                                                      framesO, MIST_P, N_FFT, NFREQ_P, 1.0f);
    }

    // 8) overlap-add + finalize
    zerof_kernel<<<ceil_div(4 * OUTLEN, B), B, 0, stream>>>(yacc, 4 * OUTLEN);
    ola_kernel<<<ceil_div(MIST * N_FFT, B), B, 0, stream>>>(framesO, yacc);
    fin_kernel<<<ceil_div(out_size, B), B, 0, stream>>>(yacc, smask, out, out_size);
}
